// STModel_66675072303846
// MI455X (gfx1250) — matrix operand order repaired and checked by reading, not compiled, not run on hardware
//
#include <hip/hip_runtime.h>
#include <hip/hip_bf16.h>
#include <math.h>

typedef __attribute__((ext_vector_type(16))) _Float16 v16h;
typedef __attribute__((ext_vector_type(8)))  _Float16 h8;
typedef __attribute__((ext_vector_type(8)))  float    v8f;

#define D_MODEL 1024
#define N_HEADS 16
#define DH      64
#define TT      2
#define HW      9
#define NTOK    18
#define BATCH   512
#define MROWS   (BATCH * NTOK)   // 9216
#define NLAYER  4
#define VOCAB   84

#define BM  128
#define BN  128
#define BK  32
#define LDA 40                    // padded LDS row stride in f16 (80 B, 16B-aligned)

// ---------------------------------------------------------------------------
// Async global -> LDS 16-byte copy (gfx1250 GLOBAL_LOAD_ASYNC_TO_LDS_B128).
// VDST carries the LDS byte offset (low 32 bits of the generic LDS pointer).
// Tracked by ASYNCcnt; consumer must s_wait_asynccnt then barrier.
// ---------------------------------------------------------------------------
__device__ __forceinline__ void async_copy16(const _Float16* gsrc, _Float16* ldst) {
    unsigned l = (unsigned)(uintptr_t)ldst;          // LDS aperture: addr[31:0] = LDS offset
    asm volatile("global_load_async_to_lds_b128 %0, %1, off"
                 :: "v"(l), "v"((const void*)gsrc)
                 : "memory");
}
__device__ __forceinline__ void wait_async_all() {
    asm volatile("s_wait_asynccnt 0" ::: "memory");
}

// ---------------------------------------------------------------------------
// Weight transpose + fp32 -> f16:  W (K x N) row-major -> Wt (N x K).
// 32x32 LDS-tiled so both global streams are coalesced.
// ---------------------------------------------------------------------------
__global__ void tconv_kernel(const float* __restrict__ W, _Float16* __restrict__ Wt,
                             int K, int N) {
    __shared__ _Float16 tile[32][33];
    int tk0 = blockIdx.x * 32;           // K tile origin
    int tn0 = blockIdx.y * 32;           // N tile origin
    int tx = threadIdx.x & 31, ty = threadIdx.x >> 5;   // 256 thr = 32 x 8
    #pragma unroll
    for (int r = ty; r < 32; r += 8)
        tile[r][tx] = (_Float16)W[(size_t)(tk0 + r) * N + tn0 + tx];
    __syncthreads();
    #pragma unroll
    for (int r = ty; r < 32; r += 8)
        Wt[(size_t)(tn0 + r) * K + tk0 + tx] = tile[tx][r];
}

// ---------------------------------------------------------------------------
// Embedding: x[b,p,:] = token_emb[seq[b,p]] + pos2d[p%9] + time_emb[p/9]
// ---------------------------------------------------------------------------
__global__ void embed_kernel(const float* __restrict__ tok,
                             const float* __restrict__ timee,
                             const float* __restrict__ pos,
                             const int*   __restrict__ seq,
                             float* __restrict__ x) {
    int idx = blockIdx.x * blockDim.x + threadIdx.x;
    if (idx >= MROWS * D_MODEL) return;
    int d = idx & (D_MODEL - 1);
    int m = idx >> 10;
    int b = m / NTOK, p = m % NTOK;
    int tk = seq[b * NTOK + p];
    x[idx] = tok[(size_t)tk * D_MODEL + d]
           + pos[(size_t)(p % HW) * D_MODEL + d]
           + timee[(size_t)(p / HW) * D_MODEL + d];
}

// ---------------------------------------------------------------------------
// LayerNorm (row = 1024), fp32 in -> f16 out (WMMA A operand)
// ---------------------------------------------------------------------------
__global__ void ln_kernel(const float* __restrict__ x,
                          const float* __restrict__ g,
                          const float* __restrict__ b,
                          _Float16* __restrict__ y) {
    __shared__ float red[256];
    int row = blockIdx.x, tid = threadIdx.x;
    const float* xr = x + (size_t)row * D_MODEL;
    float s = 0.f, s2 = 0.f;
    for (int d = tid; d < D_MODEL; d += 256) { float v = xr[d]; s += v; s2 += v * v; }
    red[tid] = s; __syncthreads();
    for (int o = 128; o; o >>= 1) { if (tid < o) red[tid] += red[tid + o]; __syncthreads(); }
    float mean = red[0] * (1.0f / D_MODEL);
    __syncthreads();
    red[tid] = s2; __syncthreads();
    for (int o = 128; o; o >>= 1) { if (tid < o) red[tid] += red[tid + o]; __syncthreads(); }
    float var = red[0] * (1.0f / D_MODEL) - mean * mean;
    float rinv = rsqrtf(var + 1e-5f);
    for (int d = tid; d < D_MODEL; d += 256)
        y[(size_t)row * D_MODEL + d] = (_Float16)((xr[d] - mean) * rinv * g[d] + b[d]);
}

// ---------------------------------------------------------------------------
// WMMA GEMM:  C[M,N] = A[M,K] (f16 row-major) * Bt[N,K]^T (f16)
// 256-thread block = 8 waves; block tile 128x128; wave tile 32(M) x 64(N).
// K-panels (128x32) double-buffered in LDS, filled with async global->LDS.
// Epilogue modes: 0 = store f16 ; 1 = resid += C + bias ; 2 = f16 gelu(C+bias)
// ---------------------------------------------------------------------------
__global__ void __launch_bounds__(256)
gemm_wmma_kernel(const _Float16* __restrict__ A,
                 const _Float16* __restrict__ Bt,
                 int M, int N, int K, int mode,
                 const float* __restrict__ bias,
                 float* __restrict__ resid,
                 _Float16* __restrict__ out16) {
    __shared__ __align__(16) _Float16 As[2][BM * LDA];
    __shared__ __align__(16) _Float16 Bs[2][BM * LDA];

    const int tid  = threadIdx.x;
    const int lane = tid & 31;
    const int w    = tid >> 5;            // 0..7
    const int wm   = w & 3;               // 4 wave-rows  * 32 M
    const int wn   = w >> 2;              // 2 wave-cols  * 64 N
    const int lo   = lane & 15;
    const int hi   = lane >> 4;

    const int bm0 = blockIdx.x * BM;
    const int bn0 = blockIdx.y * BN;

    union Frag { v16h v; h8 h[2]; };
    v8f acc[2][4];
    v8f zero = {};
    #pragma unroll
    for (int mm = 0; mm < 2; ++mm)
        #pragma unroll
        for (int s = 0; s < 4; ++s) acc[mm][s] = zero;

    const _Float16* Ag = A  + (size_t)bm0 * K;
    const _Float16* Bg = Bt + (size_t)bn0 * K;

    // stage one 128x32 panel: 512 16-byte chunks, 2 per thread
    auto stage = [&](int buf, int k0) {
        #pragma unroll
        for (int c = tid; c < 512; c += 256) {
            int row = c >> 2, kc = (c & 3) * 8;
            async_copy16(Ag + (size_t)row * K + k0 + kc, &As[buf][row * LDA + kc]);
            async_copy16(Bg + (size_t)row * K + k0 + kc, &Bs[buf][row * LDA + kc]);
        }
    };

    stage(0, 0);
    const int ksteps = K / BK;
    for (int kt = 0; kt < ksteps; ++kt) {
        int buf = kt & 1;
        wait_async_all();                 // own async loads written to LDS
        __syncthreads();                  // whole tile visible to all waves
        if (kt + 1 < ksteps) stage(buf ^ 1, (kt + 1) * BK);

        Frag a[2], b[4];
        #pragma unroll
        for (int mm = 0; mm < 2; ++mm) {
            const _Float16* ap = &As[buf][(wm * 32 + mm * 16 + lo) * LDA];
            a[mm].h[0] = *(const h8*)(ap + hi * 8);
            a[mm].h[1] = *(const h8*)(ap + 16 + hi * 8);
        }
        #pragma unroll
        for (int s = 0; s < 4; ++s) {
            const _Float16* bp = &Bs[buf][(wn * 64 + s * 16 + lo) * LDA];
            b[s].h[0] = *(const h8*)(bp + hi * 8);
            b[s].h[1] = *(const h8*)(bp + 16 + hi * 8);
        }
        #pragma unroll
        for (int mm = 0; mm < 2; ++mm)
            #pragma unroll
            for (int s = 0; s < 4; ++s)
                acc[mm][s] = __builtin_amdgcn_wmma_f32_16x16x32_f16(
                    false, a[mm].v, false, b[s].v, (short)0, acc[mm][s], false, false);

        __syncthreads();                  // all waves done with buf before overwrite
    }

    #pragma unroll
    for (int mm = 0; mm < 2; ++mm) {
        #pragma unroll
        for (int s = 0; s < 4; ++s) {
            #pragma unroll
            for (int r = 0; r < 8; ++r) {
                float val = acc[mm][s][r];
                int mg = bm0 + wm * 32 + mm * 16 + r + 8 * hi;
                int ng = bn0 + wn * 64 + s * 16 + lo;
                size_t o = (size_t)mg * N + ng;
                if (mode == 0) {
                    out16[o] = (_Float16)val;
                } else if (mode == 1) {
                    resid[o] += val + bias[ng];
                } else {
                    float u = val + bias[ng];
                    out16[o] = (_Float16)(0.5f * u * (1.0f + erff(u * 0.70710678f)));
                }
            }
        }
    }
}

// ---------------------------------------------------------------------------
// Spatial axial attention: per (b, t, head), 9 tokens x 64 dims, axial RoPE.
// qkv layout: (M, 3*1024) f16 ; out: (M, 1024) f16
// ---------------------------------------------------------------------------
__global__ void spatial_attn_kernel(const _Float16* __restrict__ qkv,
                                    _Float16* __restrict__ out) {
    const int blk  = blockIdx.x;             // b*32 + t*16 + head
    const int head = blk & 15;
    const int t    = (blk >> 4) & 1;
    const int b    = blk >> 5;
    const int tid  = threadIdx.x;            // 128
    __shared__ float q[HW][DH], k[HW][DH], v[HW][DH];
    __shared__ float sc[HW][HW], pr[HW][HW];
    const float PI = 3.14159265358979f;

    for (int it = tid; it < HW * 32; it += 128) {     // 288 interleaved pairs
        int s = it / 32, i = it % 32;
        int yy = s / 3, xx = s % 3;
        float f = (1.0f + 4.0f * (float)(i & 15) * (1.0f / 15.0f)) * PI;
        float ang = (i < 16) ? (float)(yy - 1) * f : (float)(xx - 1) * f;
        float c = cosf(ang), sn = sinf(ang);
        size_t base = ((size_t)(b * NTOK + t * HW + s)) * 3072 + head * DH + 2 * i;
        float q1 = (float)qkv[base], q2 = (float)qkv[base + 1];
        q[s][2*i] = q1 * c - q2 * sn;  q[s][2*i+1] = q1 * sn + q2 * c;
        float k1 = (float)qkv[base + 1024], k2 = (float)qkv[base + 1025];
        k[s][2*i] = k1 * c - k2 * sn;  k[s][2*i+1] = k1 * sn + k2 * c;
        v[s][2*i] = (float)qkv[base + 2048];  v[s][2*i+1] = (float)qkv[base + 2049];
    }
    __syncthreads();
    for (int it = tid; it < HW * HW; it += 128) {
        int i = it / HW, j = it % HW;
        float acc = 0.f;
        for (int d = 0; d < DH; ++d) acc += q[i][d] * k[j][d];
        sc[i][j] = acc * 0.125f;     // 1/sqrt(64)
    }
    __syncthreads();
    if (tid < HW) {
        float mx = -1e30f;
        for (int j = 0; j < HW; ++j) mx = fmaxf(mx, sc[tid][j]);
        float e[HW], sum = 0.f;
        for (int j = 0; j < HW; ++j) { e[j] = expf(sc[tid][j] - mx); sum += e[j]; }
        float inv = 1.0f / sum;
        for (int j = 0; j < HW; ++j) pr[tid][j] = e[j] * inv;
    }
    __syncthreads();
    for (int it = tid; it < HW * DH; it += 128) {
        int i = it / DH, d = it % DH;
        float acc = 0.f;
        for (int j = 0; j < HW; ++j) acc += pr[i][j] * v[j][d];
        out[((size_t)(b * NTOK + t * HW + i)) * D_MODEL + head * DH + d] = (_Float16)acc;
    }
}

// ---------------------------------------------------------------------------
// Temporal causal attention: per (b, spatial, head), T=2, temporal RoPE.
// ---------------------------------------------------------------------------
__global__ void temporal_attn_kernel(const _Float16* __restrict__ qkv,
                                     _Float16* __restrict__ out) {
    const int blk  = blockIdx.x;             // b*144 + s*16 + head
    const int head = blk & 15;
    const int s    = (blk / 16) % HW;
    const int b    = blk / 144;
    const int tid  = threadIdx.x;            // 64
    __shared__ float q[TT][DH], k[TT][DH], v[TT][DH];
    __shared__ float p0s, p1s;

    int t = tid >> 5, i = tid & 31;          // RoPE pair (t, i)
    float inv = powf(10000.0f, -(2.0f * (float)i) * (1.0f / DH));
    float ang = (float)t * inv;
    float c = cosf(ang), sn = sinf(ang);
    size_t base = ((size_t)(b * NTOK + t * HW + s)) * 3072 + head * DH + 2 * i;
    float q1 = (float)qkv[base], q2 = (float)qkv[base + 1];
    q[t][2*i] = q1 * c - q2 * sn;  q[t][2*i+1] = q1 * sn + q2 * c;
    float k1 = (float)qkv[base + 1024], k2 = (float)qkv[base + 1025];
    k[t][2*i] = k1 * c - k2 * sn;  k[t][2*i+1] = k1 * sn + k2 * c;
    v[t][2*i] = (float)qkv[base + 2048];  v[t][2*i+1] = (float)qkv[base + 2049];
    __syncthreads();
    if (tid == 0) {
        float s10 = 0.f, s11 = 0.f;
        for (int d = 0; d < DH; ++d) { s10 += q[1][d] * k[0][d]; s11 += q[1][d] * k[1][d]; }
        s10 *= 0.125f; s11 *= 0.125f;
        float m = fmaxf(s10, s11);
        float e0 = expf(s10 - m), e1 = expf(s11 - m);
        float r = 1.0f / (e0 + e1);
        p0s = e0 * r; p1s = e1 * r;
    }
    __syncthreads();
    int d = tid;                              // t=0 attends only itself (causal)
    out[((size_t)(b * NTOK + s)) * D_MODEL + head * DH + d] = (_Float16)v[0][d];
    out[((size_t)(b * NTOK + HW + s)) * D_MODEL + head * DH + d] =
        (_Float16)(p0s * v[0][d] + p1s * v[1][d]);
}

// ---------------------------------------------------------------------------
// Head projection (small: N=84) in fp32: out[m,v] = x[m,:] . head_w[:,v] + b
// ---------------------------------------------------------------------------
__global__ void head_kernel(const float* __restrict__ x,
                            const float* __restrict__ W,
                            const float* __restrict__ bias,
                            float* __restrict__ out) {
    int idx = blockIdx.x * blockDim.x + threadIdx.x;
    if (idx >= MROWS * VOCAB) return;
    int m = idx / VOCAB, v = idx % VOCAB;
    const float* xr = x + (size_t)m * D_MODEL;
    float acc = 0.f;
    for (int d = 0; d < D_MODEL; ++d) acc += xr[d] * W[(size_t)d * VOCAB + v];
    out[idx] = acc + bias[v];
}

// ---------------------------------------------------------------------------
extern "C" void kernel_launch(void* const* d_in, const int* in_sizes, int n_in,
                              void* d_out, int out_size, void* d_ws, size_t ws_size,
                              hipStream_t stream) {
    const float* token_emb = (const float*)d_in[0];
    const float* time_emb  = (const float*)d_in[1];
    const float* pos2d     = (const float*)d_in[2];
    const float* ln1_g     = (const float*)d_in[3];
    const float* ln1_b     = (const float*)d_in[4];
    const float* ln2_g     = (const float*)d_in[5];
    const float* ln2_b     = (const float*)d_in[6];
    const float* sp_qkv    = (const float*)d_in[7];
    const float* sp_out_w  = (const float*)d_in[8];
    const float* sp_out_b  = (const float*)d_in[9];
    const float* tm_qkv    = (const float*)d_in[10];
    const float* tm_out_w  = (const float*)d_in[11];
    const float* tm_out_b  = (const float*)d_in[12];
    const float* mlp_ln_g  = (const float*)d_in[13];
    const float* mlp_ln_b  = (const float*)d_in[14];
    const float* mlp_w1    = (const float*)d_in[15];
    const float* mlp_b1    = (const float*)d_in[16];
    const float* mlp_w2    = (const float*)d_in[17];
    const float* mlp_b2    = (const float*)d_in[18];
    const float* head_w    = (const float*)d_in[19];
    const float* head_b    = (const float*)d_in[20];
    const int*   seq       = (const int*)d_in[21];
    float* out = (float*)d_out;

    const int D = D_MODEL, M = MROWS, L = NLAYER;
    char* ws = (char*)d_ws;
    size_t off = 0;
    auto alloc = [&](size_t bytes) -> char* {
        char* p = ws + off;
        off += (bytes + 255) & ~(size_t)255;
        return p;
    };
    // f16 transposed weights (128 MB total -> L2 resident)
    _Float16* wt_sp_qkv = (_Float16*)alloc((size_t)L * 3 * D * D * 2);
    _Float16* wt_sp_out = (_Float16*)alloc((size_t)L * D * D * 2);
    _Float16* wt_tm_qkv = (_Float16*)alloc((size_t)L * 3 * D * D * 2);
    _Float16* wt_tm_out = (_Float16*)alloc((size_t)L * D * D * 2);
    _Float16* wt_w1     = (_Float16*)alloc((size_t)L * 4 * D * D * 2);
    _Float16* wt_w2     = (_Float16*)alloc((size_t)L * 4 * D * D * 2);
    // activations
    float*    x      = (float*)   alloc((size_t)M * D * 4);
    _Float16* y16    = (_Float16*)alloc((size_t)M * D * 2);
    _Float16* qkv16  = (_Float16*)alloc((size_t)M * 3 * D * 2);
    _Float16* att16  = (_Float16*)alloc((size_t)M * D * 2);
    _Float16* h16    = (_Float16*)alloc((size_t)M * 4 * D * 2);

    auto tconv = [&](const float* W, _Float16* Wt, int K, int N) {
        dim3 grid(K / 32, N / 32);
        tconv_kernel<<<grid, 256, 0, stream>>>(W, Wt, K, N);
    };
    auto gemm = [&](const _Float16* A, const _Float16* Bt, int N, int K, int mode,
                    const float* bias, float* resid, _Float16* o16) {
        dim3 grid(M / BM, N / BN);
        gemm_wmma_kernel<<<grid, 256, 0, stream>>>(A, Bt, M, N, K, mode, bias, resid, o16);
    };

    // One-time weight convert/transpose
    for (int l = 0; l < L; ++l) {
        tconv(sp_qkv   + (size_t)l * D * 3 * D, wt_sp_qkv + (size_t)l * 3 * D * D, D, 3 * D);
        tconv(sp_out_w + (size_t)l * D * D,     wt_sp_out + (size_t)l * D * D,     D, D);
        tconv(tm_qkv   + (size_t)l * D * 3 * D, wt_tm_qkv + (size_t)l * 3 * D * D, D, 3 * D);
        tconv(tm_out_w + (size_t)l * D * D,     wt_tm_out + (size_t)l * D * D,     D, D);
        tconv(mlp_w1   + (size_t)l * D * 4 * D, wt_w1     + (size_t)l * 4 * D * D, D, 4 * D);
        tconv(mlp_w2   + (size_t)l * 4 * D * D, wt_w2     + (size_t)l * 4 * D * D, 4 * D, D);
    }

    // Embedding
    embed_kernel<<<(M * D + 255) / 256, 256, 0, stream>>>(token_emb, time_emb, pos2d, seq, x);

    for (int l = 0; l < L; ++l) {
        // spatial attention block
        ln_kernel<<<M, 256, 0, stream>>>(x, ln1_g + l * D, ln1_b + l * D, y16);
        gemm(y16, wt_sp_qkv + (size_t)l * 3 * D * D, 3 * D, D, 0, nullptr, nullptr, qkv16);
        spatial_attn_kernel<<<BATCH * TT * N_HEADS, 128, 0, stream>>>(qkv16, att16);
        gemm(att16, wt_sp_out + (size_t)l * D * D, D, D, 1, sp_out_b + l * D, x, nullptr);
        // temporal attention block
        ln_kernel<<<M, 256, 0, stream>>>(x, ln2_g + l * D, ln2_b + l * D, y16);
        gemm(y16, wt_tm_qkv + (size_t)l * 3 * D * D, 3 * D, D, 0, nullptr, nullptr, qkv16);
        temporal_attn_kernel<<<BATCH * HW * N_HEADS, 64, 0, stream>>>(qkv16, att16);
        gemm(att16, wt_tm_out + (size_t)l * D * D, D, D, 1, tm_out_b + l * D, x, nullptr);
        // MLP block
        ln_kernel<<<M, 256, 0, stream>>>(x, mlp_ln_g + l * D, mlp_ln_b + l * D, y16);
        gemm(y16, wt_w1 + (size_t)l * 4 * D * D, 4 * D, D, 2, mlp_b1 + (size_t)l * 4 * D, nullptr, h16);
        gemm(h16, wt_w2 + (size_t)l * 4 * D * D, D, 4 * D, 1, mlp_b2 + l * D, x, nullptr);
    }

    head_kernel<<<(M * VOCAB + 255) / 256, 256, 0, stream>>>(x, head_w, head_b, out);
}
